// EncSwitchedFC_34187939676855
// MI455X (gfx1250) — matrix operand order repaired and checked by reading, not compiled, not run on hardware
//
#include <hip/hip_runtime.h>
#include <hip/hip_bf16.h>
#include <math.h>
#include <stdint.h>

#define B  16384
#define D  1024
#define H  256
#define NB 16
#define CTRL 48  // 3*NB
#define TILES_PER_EXP (B / 16)  // 1024

typedef __attribute__((ext_vector_type(16))) __bf16 v16bf;
typedef __attribute__((ext_vector_type(8)))  __bf16 v8bf;
typedef __attribute__((ext_vector_type(4)))  __bf16 v4bf;
typedef __attribute__((ext_vector_type(8)))  float  v8f;
typedef __attribute__((ext_vector_type(4)))  float  v4f;
typedef __attribute__((ext_vector_type(4)))  int    v4i;

// Output tuple layout (flat, f32): x+out | y_logits | y_idx | y | z_mean | z_logvar | z
constexpr size_t OFF_YL   = (size_t)B * D;
constexpr size_t OFF_YIDX = OFF_YL + (size_t)B * NB;
constexpr size_t OFF_Y    = OFF_YIDX + (size_t)B;
constexpr size_t OFF_ZM   = OFF_Y + (size_t)B * NB;
constexpr size_t OFF_LV   = OFF_ZM + (size_t)B * NB;
constexpr size_t OFF_Z    = OFF_LV + (size_t)B * NB;

static __device__ __forceinline__ v16bf cat8(v8bf lo, v8bf hi) {
  return __builtin_shufflevector(lo, hi, 0,1,2,3,4,5,6,7,8,9,10,11,12,13,14,15);
}

// ---- CDNA5 async global->LDS copy (ASYNCcnt-tracked), builtin if present ----

typedef __attribute__((address_space(1))) v4i g_v4i;
typedef __attribute__((address_space(3))) v4i l_v4i;

static __device__ __forceinline__ void async_copy_b128(const void* g, void* l) {
#if __has_builtin(__builtin_amdgcn_global_load_async_to_lds_b128)
  __builtin_amdgcn_global_load_async_to_lds_b128(
      (g_v4i*)(void*)g, (l_v4i*)l, 0, 0);
#else
  unsigned loff = (unsigned)(uintptr_t)(__attribute__((address_space(3))) void*)l;
  asm volatile("global_load_async_to_lds_b128 %0, %1, off"
               :: "v"(loff), "v"(g) : "memory");
#endif
}

static __device__ __forceinline__ void wait_async0() {
#if __has_builtin(__builtin_amdgcn_s_wait_asynccnt)
  __builtin_amdgcn_s_wait_asynccnt(0);
#else
  asm volatile("s_wait_asynccnt 0" ::: "memory");
#endif
}

// ---------------- precision / layout conversion ----------------

__global__ __launch_bounds__(256) void k_convert_x(const float* __restrict__ x,
                                                   __bf16* __restrict__ xbf) {
  size_t i = ((size_t)blockIdx.x * 256 + threadIdx.x) * 4;
  v4f v = *(const v4f*)(x + i);
  v4bf o;
  o[0] = (__bf16)v[0]; o[1] = (__bf16)v[1]; o[2] = (__bf16)v[2]; o[3] = (__bf16)v[3];
  *(v4bf*)(xbf + i) = o;
}

// src slab: R x C f32 (row-major) -> dst slab: C x R bf16 (row-major)
__global__ __launch_bounds__(256) void k_transpose(const float* __restrict__ src,
                                                   __bf16* __restrict__ dst,
                                                   int R, int C) {
  __shared__ float tile[32][33];
  size_t slab = (size_t)blockIdx.z * R * C;
  src += slab; dst += slab;
  int c0 = blockIdx.x * 32, r0 = blockIdx.y * 32;
  int tx = threadIdx.x & 31, ty = threadIdx.x >> 5;  // 32 x 8
  #pragma unroll
  for (int i = 0; i < 4; ++i)
    tile[ty + 8 * i][tx] = src[(size_t)(r0 + ty + 8 * i) * C + c0 + tx];
  __syncthreads();
  #pragma unroll
  for (int i = 0; i < 4; ++i)
    dst[(size_t)(c0 + ty + 8 * i) * R + r0 + tx] = (__bf16)tile[tx][ty + 8 * i];
}

__global__ void k_zero_counts(int* __restrict__ counts) {
  if (threadIdx.x < NB) counts[threadIdx.x] = 0;
}

// ---------------- GEMM1: ctrl_h = relu(x @ Ws1 + bs1)  (B x H) ----------------

__global__ __launch_bounds__(256) void k_gemm1(const __bf16* __restrict__ xbf,
                                               const __bf16* __restrict__ ws1t,  // H x D
                                               const float* __restrict__ bs1,
                                               float* __restrict__ ctrlh) {
  __shared__ __bf16 abuf[16][64];  // 16 rows x 64 K, staged async
  int tid = threadIdx.x;
  int w = tid >> 5, l = tid & 31, half = l >> 4, ln = l & 15;
  int m0 = blockIdx.x * 16;
  int c0 = w * 32;
  const __bf16* b0p = ws1t + (size_t)(c0 + ln) * D;
  const __bf16* b1p = ws1t + (size_t)(c0 + 16 + ln) * D;
  int srow = tid >> 3, schunk = (tid & 7) * 8;   // stage mapping (tid < 128)
  const __bf16* sgp = xbf + (size_t)(m0 + srow) * D + schunk;
  v8f acc0 = {}, acc1 = {};
  for (int k0 = 0; k0 < D; k0 += 64) {
    __syncthreads();                       // prior iteration's LDS reads done
    if (tid < 128) async_copy_b128(sgp + k0, &abuf[srow][schunk]);
    wait_async0();
    __syncthreads();                       // tile visible to all waves
    #pragma unroll
    for (int sub = 0; sub < 64; sub += 32) {
      v16bf a = cat8(*(const v8bf*)&abuf[ln][sub + 16 * half],
                     *(const v8bf*)&abuf[ln][sub + 16 * half + 8]);
      int kb = k0 + sub + 16 * half;
      v16bf b0 = cat8(*(const v8bf*)(b0p + kb), *(const v8bf*)(b0p + kb + 8));
      v16bf b1 = cat8(*(const v8bf*)(b1p + kb), *(const v8bf*)(b1p + kb + 8));
      acc0 = __builtin_amdgcn_wmma_f32_16x16x32_bf16(false, a, false, b0, (short)0, acc0, false, false);
      acc1 = __builtin_amdgcn_wmma_f32_16x16x32_bf16(false, a, false, b1, (short)0, acc1, false, false);
    }
  }
  float bi0 = bs1[c0 + ln], bi1 = bs1[c0 + 16 + ln];
  #pragma unroll
  for (int v = 0; v < 8; ++v) {
    int m = m0 + v + 8 * half;
    ctrlh[(size_t)m * H + c0 + ln]      = fmaxf(acc0[v] + bi0, 0.f);
    ctrlh[(size_t)m * H + c0 + 16 + ln] = fmaxf(acc1[v] + bi1, 0.f);
  }
}

// ------- ctrl = ctrl_h @ Ws2 + bs2 ; gumbel softmax; z; bucket rows by expert -------

__global__ __launch_bounds__(64) void k_ctrl(const float* __restrict__ ctrlh,
                                             const float* __restrict__ Ws2,
                                             const float* __restrict__ bs2,
                                             const float* __restrict__ u,
                                             const float* __restrict__ eps,
                                             float* __restrict__ out,
                                             float* __restrict__ zsel,
                                             int* __restrict__ counts,
                                             int* __restrict__ bucket) {
  __shared__ float sh[H];
  __shared__ float c[CTRL];
  __shared__ float av[NB], ev[NB], zv[NB];
  __shared__ float mx, ssum;
  __shared__ int mi;
  int b = blockIdx.x, tid = threadIdx.x;
  for (int i = tid; i < H; i += 64) sh[i] = ctrlh[(size_t)b * H + i];
  __syncthreads();
  if (tid < CTRL) {
    float s = bs2[tid];
    for (int k = 0; k < H; ++k) s = fmaf(sh[k], Ws2[k * CTRL + tid], s);
    c[tid] = s;
  }
  __syncthreads();
  if (tid < NB) {
    float g = -logf(-logf(u[(size_t)b * NB + tid]));
    av[tid] = c[tid] + g;
    zv[tid] = c[NB + tid] + expf(0.5f * c[2 * NB + tid]) * eps[(size_t)b * NB + tid];
  }
  __syncthreads();
  if (tid == 0) {
    float m = av[0]; int im = 0;
    for (int i = 1; i < NB; ++i) if (av[i] > m) { m = av[i]; im = i; }
    mx = m; mi = im;
  }
  __syncthreads();
  if (tid < NB) ev[tid] = expf(av[tid] - mx);
  __syncthreads();
  if (tid == 0) {
    float s = 0.f;
    for (int i = 0; i < NB; ++i) s += ev[i];
    ssum = s;
  }
  __syncthreads();
  if (tid < NB) {
    size_t o = (size_t)b * NB + tid;
    out[OFF_YL + o] = c[tid];
    out[OFF_Y  + o] = ev[tid] / ssum;
    out[OFF_ZM + o] = c[NB + tid];
    out[OFF_LV + o] = c[2 * NB + tid];
    out[OFF_Z  + o] = zv[tid];
  }
  if (tid == 0) {
    out[OFF_YIDX + b] = (float)mi;
    zsel[b] = zv[mi];
    int pos = atomicAdd(&counts[mi], 1);
    bucket[mi * B + pos] = b;
  }
}

// ------- expert GEMM 1: h = relu(x[rows] @ W1[n] + b1[n]) * zsel[rows]  -> bf16 -------

__global__ __launch_bounds__(256) void k_exp1(const __bf16* __restrict__ xbf,
                                              const __bf16* __restrict__ w1t,  // NB x H x D
                                              const float* __restrict__ b1,
                                              const float* __restrict__ zsel,
                                              const int* __restrict__ counts,
                                              const int* __restrict__ bucket,
                                              __bf16* __restrict__ hbf) {
  __shared__ int rows[16];
  __shared__ float zs[16];
  __shared__ int cnt_s;
  __shared__ __bf16 abuf[16][64];
  int tid = threadIdx.x;
  int n = blockIdx.x >> 10;        // TILES_PER_EXP == 1024
  int t = blockIdx.x & (TILES_PER_EXP - 1);
  if (tid == 0) cnt_s = counts[n];
  __syncthreads();
  int cnt = cnt_s;
  if (t * 16 >= cnt) return;       // block-uniform: EXEC stays full for WMMA
  if (tid < 16) {
    int r = t * 16 + tid;
    int rc = r < cnt ? r : cnt - 1;
    int row = bucket[n * B + rc];
    rows[tid] = row;
    zs[tid] = zsel[row];
  }
  __syncthreads();
  int w = tid >> 5, l = tid & 31, half = l >> 4, ln = l & 15;
  int c0 = w * 32;
  const __bf16* wb  = w1t + (size_t)n * H * D;
  const __bf16* b0p = wb + (size_t)(c0 + ln) * D;
  const __bf16* b1p = wb + (size_t)(c0 + 16 + ln) * D;
  int srow = tid >> 3, schunk = (tid & 7) * 8;
  const __bf16* sgp = (tid < 128) ? xbf + (size_t)rows[srow] * D + schunk : xbf;
  v8f acc0 = {}, acc1 = {};
  for (int k0 = 0; k0 < D; k0 += 64) {
    __syncthreads();
    if (tid < 128) async_copy_b128(sgp + k0, &abuf[srow][schunk]);
    wait_async0();
    __syncthreads();
    #pragma unroll
    for (int sub = 0; sub < 64; sub += 32) {
      v16bf a = cat8(*(const v8bf*)&abuf[ln][sub + 16 * half],
                     *(const v8bf*)&abuf[ln][sub + 16 * half + 8]);
      int kb = k0 + sub + 16 * half;
      v16bf bf0 = cat8(*(const v8bf*)(b0p + kb), *(const v8bf*)(b0p + kb + 8));
      v16bf bf1 = cat8(*(const v8bf*)(b1p + kb), *(const v8bf*)(b1p + kb + 8));
      acc0 = __builtin_amdgcn_wmma_f32_16x16x32_bf16(false, a, false, bf0, (short)0, acc0, false, false);
      acc1 = __builtin_amdgcn_wmma_f32_16x16x32_bf16(false, a, false, bf1, (short)0, acc1, false, false);
    }
  }
  float bi0 = b1[n * H + c0 + ln], bi1 = b1[n * H + c0 + 16 + ln];
  #pragma unroll
  for (int v = 0; v < 8; ++v) {
    int m = v + 8 * half;
    if (t * 16 + m < cnt) {
      int row = rows[m];
      float zz = zs[m];
      hbf[(size_t)row * H + c0 + ln]      = (__bf16)(fmaxf(acc0[v] + bi0, 0.f) * zz);
      hbf[(size_t)row * H + c0 + 16 + ln] = (__bf16)(fmaxf(acc1[v] + bi1, 0.f) * zz);
    }
  }
}

// ------- expert GEMM 2: out = x[rows] + h[rows] @ W2[n] + b2[n] -------

__global__ __launch_bounds__(256) void k_exp2(const __bf16* __restrict__ hbf,
                                              const __bf16* __restrict__ w2t,  // NB x D x H
                                              const float* __restrict__ b2,
                                              const float* __restrict__ x,
                                              const int* __restrict__ counts,
                                              const int* __restrict__ bucket,
                                              float* __restrict__ out0) {
  __shared__ int rows[16];
  __shared__ int cnt_s;
  __shared__ __bf16 abuf[16][64];
  int tid = threadIdx.x;
  int n = blockIdx.x >> 10;
  int t = blockIdx.x & (TILES_PER_EXP - 1);
  if (tid == 0) cnt_s = counts[n];
  __syncthreads();
  int cnt = cnt_s;
  if (t * 16 >= cnt) return;
  if (tid < 16) {
    int r = t * 16 + tid;
    int rc = r < cnt ? r : cnt - 1;
    rows[tid] = bucket[n * B + rc];
  }
  __syncthreads();
  int w = tid >> 5, l = tid & 31, half = l >> 4, ln = l & 15;
  int c0 = blockIdx.y * 256 + w * 32;  // grid.y = 4 -> covers D=1024 columns
  const __bf16* wb  = w2t + (size_t)n * D * H;
  const __bf16* b0p = wb + (size_t)(c0 + ln) * H;
  const __bf16* b1p = wb + (size_t)(c0 + 16 + ln) * H;
  int srow = tid >> 3, schunk = (tid & 7) * 8;
  const __bf16* sgp = (tid < 128) ? hbf + (size_t)rows[srow] * H + schunk : hbf;
  v8f acc0 = {}, acc1 = {};
  for (int k0 = 0; k0 < H; k0 += 64) {
    __syncthreads();
    if (tid < 128) async_copy_b128(sgp + k0, &abuf[srow][schunk]);
    wait_async0();
    __syncthreads();
    #pragma unroll
    for (int sub = 0; sub < 64; sub += 32) {
      v16bf a = cat8(*(const v8bf*)&abuf[ln][sub + 16 * half],
                     *(const v8bf*)&abuf[ln][sub + 16 * half + 8]);
      int kb = k0 + sub + 16 * half;
      v16bf bf0 = cat8(*(const v8bf*)(b0p + kb), *(const v8bf*)(b0p + kb + 8));
      v16bf bf1 = cat8(*(const v8bf*)(b1p + kb), *(const v8bf*)(b1p + kb + 8));
      acc0 = __builtin_amdgcn_wmma_f32_16x16x32_bf16(false, a, false, bf0, (short)0, acc0, false, false);
      acc1 = __builtin_amdgcn_wmma_f32_16x16x32_bf16(false, a, false, bf1, (short)0, acc1, false, false);
    }
  }
  float bi0 = b2[n * D + c0 + ln], bi1 = b2[n * D + c0 + 16 + ln];
  #pragma unroll
  for (int v = 0; v < 8; ++v) {
    int m = v + 8 * half;
    if (t * 16 + m < cnt) {
      int row = rows[m];
      size_t o0 = (size_t)row * D + c0 + ln;
      size_t o1 = (size_t)row * D + c0 + 16 + ln;
      out0[o0] = x[o0] + acc0[v] + bi0;
      out0[o1] = x[o1] + acc1[v] + bi1;
    }
  }
}

// ---------------- host launcher ----------------

extern "C" void kernel_launch(void* const* d_in, const int* in_sizes, int n_in,
                              void* d_out, int out_size, void* d_ws, size_t ws_size,
                              hipStream_t stream) {
  (void)in_sizes; (void)n_in; (void)out_size; (void)ws_size;
  const float* x   = (const float*)d_in[0];
  const float* Ws1 = (const float*)d_in[1];
  const float* bs1 = (const float*)d_in[2];
  const float* Ws2 = (const float*)d_in[3];
  const float* bs2 = (const float*)d_in[4];
  const float* W1  = (const float*)d_in[5];
  const float* b1  = (const float*)d_in[6];
  const float* W2  = (const float*)d_in[7];
  const float* b2  = (const float*)d_in[8];
  const float* u   = (const float*)d_in[9];
  const float* eps = (const float*)d_in[10];
  float* out = (float*)d_out;

  char* p = (char*)d_ws;
  __bf16* xbf  = (__bf16*)p; p += (size_t)B * D * 2;
  __bf16* ws1t = (__bf16*)p; p += (size_t)D * H * 2;
  __bf16* w1t  = (__bf16*)p; p += (size_t)NB * D * H * 2;
  __bf16* w2t  = (__bf16*)p; p += (size_t)NB * D * H * 2;
  float* ctrlh = (float*)p;  p += (size_t)B * H * 4;
  __bf16* hbf  = (__bf16*)p; p += (size_t)B * H * 2;
  float* zsel  = (float*)p;  p += (size_t)B * 4;
  int* counts  = (int*)p;    p += 256;
  int* bucket  = (int*)p;    p += (size_t)NB * B * 4;

  k_convert_x<<<(B * D) / 1024, 256, 0, stream>>>(x, xbf);
  k_transpose<<<dim3(H / 32, D / 32, 1), 256, 0, stream>>>(Ws1, ws1t, D, H);
  k_transpose<<<dim3(H / 32, D / 32, NB), 256, 0, stream>>>(W1, w1t, D, H);
  k_transpose<<<dim3(D / 32, H / 32, NB), 256, 0, stream>>>(W2, w2t, H, D);
  k_zero_counts<<<1, 32, 0, stream>>>(counts);
  k_gemm1<<<B / 16, 256, 0, stream>>>(xbf, ws1t, bs1, ctrlh);
  k_ctrl<<<B, 64, 0, stream>>>(ctrlh, Ws2, bs2, u, eps, out, zsel, counts, bucket);
  k_exp1<<<NB * TILES_PER_EXP, 256, 0, stream>>>(xbf, w1t, b1, zsel, counts, bucket, hbf);
  k_exp2<<<dim3(NB * TILES_PER_EXP, 4), 256, 0, stream>>>(hbf, w2t, b2, x, counts, bucket, out);
}
